// FocalSparseBEVBackBone_77988016160817
// MI455X (gfx1250) — matrix operand order repaired and checked by reading, not compiled, not run on hardware
//
#include <hip/hip_runtime.h>

// ---------------------------------------------------------------------------
// CDNA5 (gfx1250) FocalSparseBEVBackBone forward.
// Every conv = implicit GEMM on v_wmma_f32_16x16x32_f16 (wave32).
// - wave tile M=32 x N=16: 2 WMMAs/chunk sharing one B fragment
// - double-buffered LDS pipeline: chunk c+1's A gather (VMEM) and B slab
//   (Tensor Data Mover, TENSORcnt) overlap chunk c's WMMAs; one barrier/chunk
// - weights pre-packed chunk-transposed [K/32][Cout][32] f16; TDM HW padding
//   (16 dwords data + 4 dwords pad) produces 80B LDS rows -> aligned,
//   bank-conflict-free ds_load_b128 fragment reads
// - Cin>=32 layers: 32-wide K chunk sits inside one tap -> vectorized b64
//   row gather, no per-element coordinate math (all extents are pow2;
//   Cin 5 is zero-padded to 8 in activations and packed weights)
// ---------------------------------------------------------------------------

typedef _Float16 f16;
typedef __attribute__((ext_vector_type(4)))  _Float16 v4h;
typedef __attribute__((ext_vector_type(8)))  _Float16 v8h;
typedef __attribute__((ext_vector_type(16))) _Float16 v16h;
typedef __attribute__((ext_vector_type(8)))  float    v8f;
typedef __attribute__((ext_vector_type(4)))  unsigned int u32x4;
typedef __attribute__((ext_vector_type(8)))  int          i32x8;
typedef __attribute__((ext_vector_type(4)))  int          i32x4;

#if defined(__has_builtin)
#if __has_builtin(__builtin_amdgcn_tensor_load_to_lds) && __has_builtin(__builtin_amdgcn_s_wait_tensorcnt)
#define HAVE_TDM 1
#endif
#endif
#ifndef HAVE_TDM
#define HAVE_TDM 0
#endif

#if HAVE_TDM
#pragma message("CDNA5: Tensor-Data-Mover builtin path ENABLED")
#else
#pragma message("CDNA5: TDM builtin not available - cooperative LDS fallback")
#endif

#define BROW 40  // padded LDS row (80 B): 16B-aligned, 20-bank stride

#if HAVE_TDM
// DMA a contiguous [rows x 32]-f16 slab Global->LDS; HW pads each row of
// 16 dwords with 4 dwords so LDS rows land on the 80-byte stride.
__device__ __forceinline__ void tdm_load_slab_f16(unsigned int lds_off,
                                                  const void* gptr,
                                                  unsigned int rows) {
  unsigned long long ga = (unsigned long long)gptr;
  u32x4 g0;
  g0[0] = 1u;                                    // count=1 valid user D#
  g0[1] = lds_off;                               // lds_addr
  g0[2] = (unsigned int)(ga & 0xFFFFFFFFu);
  g0[3] = (unsigned int)((ga >> 32) & 0x01FFFFFFu) | (2u << 30); // type=2
  i32x8 g1;
  // data_size=2B | pad_enable | pad_interval=16dw (code 3) | pad_amount=4dw (code 3)
  g1[0] = (int)((1u << 16) | (1u << 20) | (3u << 22) | (3u << 25));
  g1[1] = (int)(32u << 16);                      // tensor_dim0 = 32
  g1[2] = (int)((rows & 0xFFFFu) << 16);         // tensor_dim1 = rows
  g1[3] = (int)(((rows >> 16) & 0xFFFFu) | (32u << 16)); // tile_dim0 = 32
  g1[4] = (int)(rows & 0xFFFFu);                 // tile_dim1 = rows
  g1[5] = 32;                                    // tensor_dim0_stride = 32
  g1[6] = 0;
  g1[7] = 0;
  i32x4 z4 = {0, 0, 0, 0};
#if defined(__clang_major__) && (__clang_major__ >= 23)
  i32x8 z8 = {0, 0, 0, 0, 0, 0, 0, 0};
  __builtin_amdgcn_tensor_load_to_lds(g0, g1, z4, z4, z8, 0);
#else
  __builtin_amdgcn_tensor_load_to_lds(g0, g1, z4, z4, 0);
#endif
}
#endif

// ---------------------------------------------------------------------------
// Implicit-GEMM conv (kd=3 3D / kd=1 2D, khw fixed 3, pad_hw fixed 1).
// x:  [Di,Hi,Wi,Cinp] f16 channel-last (Cinp = pow2)
// wk: chunk-transposed [Kpad/32][Cout][32] f16
// epilogue: y = mask * relu( acc*s[n] + t[n] + res )
// ---------------------------------------------------------------------------
__global__ void conv_wmma_kernel(const f16* __restrict__ x,
                                 const f16* __restrict__ wk,
                                 const float* __restrict__ sc,
                                 const float* __restrict__ sh,
                                 const f16* __restrict__ res,
                                 const unsigned char* __restrict__ mask,
                                 f16* __restrict__ y,
                                 int Cinp, int cinS, int Cout, int Kpad, int K,
                                 int Di, int Hi, int Wi, int M,
                                 int woS, int hoS,
                                 int kd, int stride, int pad_d, int relu) {
  __shared__ f16 As[2][32][BROW];     // double-buffered 32x32 im2col tiles
  __shared__ f16 Bs[2][128 * BROW];   // double-buffered weight slabs
  const int tid    = threadIdx.x;
  const int lane   = tid & 31;
  const int wv     = tid >> 5;
  const int nwaves = blockDim.x >> 5;
  const int BN     = nwaves * 16;
  const int m0     = blockIdx.x * 32;
  const int n0blk  = blockIdx.y * BN;
  const int wMask  = (1 << woS) - 1;
  const int hMask  = (1 << hoS) - 1;
  const int NC     = Kpad >> 5;

  // ---- stage chunk `cidx` of A into buffer pb ----------------------------
  auto stageA = [&](int cidx, int pb) {
    const int k0 = cidx << 5;
    if (Cinp >= 32) {
      // fast path: whole chunk inside one tap; contiguous channels.
      const int tap = k0 >> cinS;
      const int cb  = k0 & (Cinp - 1);
      const int kx = tap % 3, t2 = tap / 3, ky = t2 % 3, kz = t2 / 3;
      for (int e = tid; e < 32 * 8; e += blockDim.x) {
        const int mm = e >> 3, seg = e & 7;
        const int gm = m0 + mm;
        v4h v = {(f16)0.f, (f16)0.f, (f16)0.f, (f16)0.f};
        if (gm < M) {
          const int xo = gm & wMask;
          const int yo = (gm >> woS) & hMask;
          const int zo = gm >> (woS + hoS);
          const int zi = zo * stride + kz - pad_d;
          const int yi = yo * stride + ky - 1;
          const int xi = xo * stride + kx - 1;
          if ((unsigned)zi < (unsigned)Di && (unsigned)yi < (unsigned)Hi &&
              (unsigned)xi < (unsigned)Wi)
            v = *(const v4h*)&x[((((size_t)zi * Hi) + yi) * Wi + xi) * Cinp +
                                cb + seg * 4];
        }
        *(v4h*)&As[pb][mm][seg * 4] = v;
      }
    } else {
      // generic path (Cinp = 8 or 16): per-element shift/mask decomposition.
      for (int e = tid; e < 32 * 32; e += blockDim.x) {
        const int mm = e >> 5, kk = e & 31;
        const int k  = k0 + kk;
        const int gm = m0 + mm;
        f16 v = (f16)0.f;
        if (k < K && gm < M) {
          const int tap = k >> cinS;
          const int c   = k & (Cinp - 1);
          const int kx = tap % 3, t2 = tap / 3, ky = t2 % 3, kz = t2 / 3;
          const int xo = gm & wMask;
          const int yo = (gm >> woS) & hMask;
          const int zo = gm >> (woS + hoS);
          const int zi = zo * stride + kz - pad_d;
          const int yi = yo * stride + ky - 1;
          const int xi = xo * stride + kx - 1;
          if ((unsigned)zi < (unsigned)Di && (unsigned)yi < (unsigned)Hi &&
              (unsigned)xi < (unsigned)Wi)
            v = x[((((size_t)zi * Hi) + yi) * Wi + xi) * Cinp + c];
        }
        As[pb][mm][kk] = v;
      }
    }
  };
  // ---- stage chunk `cidx` of B into buffer pb ----------------------------
  auto stageB = [&](int cidx, int pb) {
    const size_t bbase = ((size_t)cidx * Cout + n0blk) * 32;
#if HAVE_TDM
    if (tid == 0)
      tdm_load_slab_f16((unsigned int)(size_t)(&Bs[pb][0]), wk + bbase,
                        (unsigned)BN);
#else
    for (int e = tid; e < BN * 4; e += blockDim.x) {
      const int nn = e >> 2, sg = e & 3;
      *(v8h*)&Bs[pb][nn * BROW + sg * 8] =
          *(const v8h*)&wk[bbase + (size_t)nn * 32 + sg * 8];
    }
#endif
  };

  v8f acc0 = {0.f, 0.f, 0.f, 0.f, 0.f, 0.f, 0.f, 0.f};
  v8f acc1 = acc0;

  // prologue: stage chunk 0
  stageA(0, 0);
  stageB(0, 0);
#if HAVE_TDM
  if (tid < 32) __builtin_amdgcn_s_wait_tensorcnt((short)0);
#endif
  __syncthreads();

  for (int c = 0; c < NC; ++c) {
    const int p = c & 1;
    // ---- fragments (ISA 7.12.2 layouts), aligned b128 LDS reads ----------
    const int grp = lane >> 4;          // A: K 0-7/16-23 vs 8-15/24-31
    const int mr  = lane & 15;
    const v8h a0l = *(const v8h*)&As[p][mr][grp * 8];
    const v8h a0h = *(const v8h*)&As[p][mr][16 + grp * 8];
    const v8h a1l = *(const v8h*)&As[p][16 + mr][grp * 8];
    const v8h a1h = *(const v8h*)&As[p][16 + mr][16 + grp * 8];
    const int ncol = wv * 16 + (lane & 15);
    const int kg   = lane >> 4;         // B: K 0-7/16-23 vs 8-15/24-31
    const v8h bl = *(const v8h*)&Bs[p][ncol * BROW + kg * 8];
    const v8h bh = *(const v8h*)&Bs[p][ncol * BROW + 16 + kg * 8];
    const v16h a0 = __builtin_shufflevector(a0l, a0h, 0, 1, 2, 3, 4, 5, 6, 7,
                                            8, 9, 10, 11, 12, 13, 14, 15);
    const v16h a1 = __builtin_shufflevector(a1l, a1h, 0, 1, 2, 3, 4, 5, 6, 7,
                                            8, 9, 10, 11, 12, 13, 14, 15);
    const v16h b  = __builtin_shufflevector(bl, bh, 0, 1, 2, 3, 4, 5, 6, 7,
                                            8, 9, 10, 11, 12, 13, 14, 15);
    acc0 = __builtin_amdgcn_wmma_f32_16x16x32_f16(false, a0, false, b,
                                                  (short)0, acc0, false, false);
    acc1 = __builtin_amdgcn_wmma_f32_16x16x32_f16(false, a1, false, b,
                                                  (short)0, acc1, false, false);
    // ---- overlap: stage chunk c+1 into the other buffer while XDL runs ---
    if (c + 1 < NC) {
      stageA(c + 1, 1 - p);
      stageB(c + 1, 1 - p);
      if (c + 2 < NC)  // warm chunk c+2's slab into L2 (global_prefetch_b8)
        __builtin_prefetch(wk + ((size_t)(c + 2) * Cout + n0blk) * 32, 0, 0);
#if HAVE_TDM
      if (tid < 32) __builtin_amdgcn_s_wait_tensorcnt((short)0);
#endif
    }
    __syncthreads();
  }

  // ---- epilogue: folded BN + residual + ReLU + sparse mask ----------------
  const int n = n0blk + wv * 16 + (lane & 15);
  if (n < Cout) {
    const float s = sc[n], t = sh[n];
    const int mbase = (lane >> 4) * 8;  // lanes 16-31 -> rows 8..15 of tile
#pragma unroll
    for (int r = 0; r < 8; ++r) {
      const int gm = m0 + mbase + r;
      if (gm < M) {
        float v = acc0[r] * s + t;
        if (res) v += (float)res[(size_t)gm * Cout + n];
        if (relu) v = v > 0.f ? v : 0.f;
        if (mask && !mask[gm]) v = 0.f;
        y[(size_t)gm * Cout + n] = (f16)v;
      }
      const int gm2 = gm + 16;
      if (gm2 < M) {
        float v = acc1[r] * s + t;
        if (res) v += (float)res[(size_t)gm2 * Cout + n];
        if (relu) v = v > 0.f ? v : 0.f;
        if (mask && !mask[gm2]) v = 0.f;
        y[(size_t)gm2 * Cout + n] = (f16)v;
      }
    }
  }
}

// ---------------------------------------------------------------------------
// Helper kernels
// ---------------------------------------------------------------------------
// pack f32 (O,Ireal,taps) -> f16 chunk-transposed [Kpad/32][O][32],
// K index = tap*Cinp + i (i >= Ireal and k >= K are zero padding).
__global__ void pack_w_kernel(const float* __restrict__ w, f16* __restrict__ out,
                              int O, int Ireal, int Cinp, int taps, int Kpad) {
  const int e = blockIdx.x * blockDim.x + threadIdx.x;
  if (e >= Kpad * O) return;
  const int kk = e & 31;
  const int t  = e >> 5;
  const int o  = t % O;
  const int chunk = t / O;
  const int k  = chunk * 32 + kk;
  f16 v = (f16)0.f;
  const int K = Cinp * taps;
  if (k < K) {
    const int tap = k / Cinp, i = k - tap * Cinp;
    if (i < Ireal) v = (f16)w[((size_t)o * Ireal + i) * taps + tap];
  }
  out[((size_t)chunk * O + o) * 32 + kk] = v;
}

__global__ void fold_bn_kernel(const float* g, const float* b, const float* m,
                               const float* v, const float* bias,
                               float* s, float* t, int C) {
  const int c = blockIdx.x * blockDim.x + threadIdx.x;
  if (c >= C) return;
  const float scv = g[c] * rsqrtf(v[c] + 1e-3f);
  const float bb  = bias ? bias[c] : 0.f;
  s[c] = scv;
  t[c] = (bb - m[c]) * scv + b[c];
}

__global__ void ident_bias_kernel(const float* b, float* s, float* t, int C) {
  const int c = blockIdx.x * blockDim.x + threadIdx.x;
  if (c >= C) return;
  s[c] = 1.f;
  t[c] = b[c];
}

__global__ void scatter_kernel(const float* __restrict__ vf,
                               const int* __restrict__ vc,
                               f16* __restrict__ X0,
                               unsigned char* __restrict__ M0, int n) {
  const int i = blockIdx.x * blockDim.x + threadIdx.x;
  if (i >= n) return;
  const int z = vc[i * 4 + 1], y = vc[i * 4 + 2], x = vc[i * 4 + 3];
  const size_t base = (((size_t)z * 256 + y) * 256 + x);
  for (int c = 0; c < 5; ++c) X0[base * 8 + c] = (f16)vf[i * 5 + c];
  M0[base] = 1;
}

__global__ void dilate3d_kernel(const unsigned char* mi, unsigned char* mo,
                                int Di, int Hi, int Wi, int Do_, int Ho, int Wo) {
  const int gm = blockIdx.x * blockDim.x + threadIdx.x;
  if (gm >= Do_ * Ho * Wo) return;
  const int xo = gm % Wo, t = gm / Wo, yo = t % Ho, zo = t / Ho;
  unsigned char r = 0;
  for (int kz = 0; kz < 3; ++kz)
    for (int ky = 0; ky < 3; ++ky)
      for (int kx = 0; kx < 3; ++kx) {
        const int zi = zo * 2 + kz - 1, yi = yo * 2 + ky - 1, xi = xo * 2 + kx - 1;
        if ((unsigned)zi < (unsigned)Di && (unsigned)yi < (unsigned)Hi &&
            (unsigned)xi < (unsigned)Wi)
          r |= mi[((size_t)zi * Hi + yi) * Wi + xi];
      }
  mo[gm] = r ? 1 : 0;
}

__global__ void bev_kernel(const f16* x4, const f16* x5, const f16* x6, f16* bev) {
  const int e = blockIdx.x * blockDim.x + threadIdx.x;
  if (e >= 32 * 32 * 128) return;
  const int c = e & 127, p = e >> 7, xx = p & 31, yy = p >> 5;
  float v = 0.f;
  for (int z = 0; z < 3; ++z) v += (float)x4[(((size_t)z * 32 + yy) * 32 + xx) * 128 + c];
  if (!(yy & 1) && !(xx & 1))
    for (int z = 0; z < 2; ++z)
      v += (float)x5[(((size_t)z * 16 + (yy >> 1)) * 16 + (xx >> 1)) * 128 + c];
  if (!(yy & 3) && !(xx & 3))
    v += (float)x6[((size_t)(yy >> 2) * 8 + (xx >> 2)) * 128 + c];
  bev[(size_t)p * 128 + c] = (f16)v;
}

__global__ void bev_mask_kernel(const unsigned char* m4, const unsigned char* m5,
                                const unsigned char* m6, unsigned char* mb) {
  const int p = blockIdx.x * blockDim.x + threadIdx.x;
  if (p >= 1024) return;
  const int xx = p & 31, yy = p >> 5;
  unsigned char r = 0;
  for (int z = 0; z < 3; ++z) r |= m4[(z * 32 + yy) * 32 + xx];
  if (!(yy & 1) && !(xx & 1))
    for (int z = 0; z < 2; ++z) r |= m5[(z * 16 + (yy >> 1)) * 16 + (xx >> 1)];
  if (!(yy & 3) && !(xx & 3)) r |= m6[(yy >> 2) * 8 + (xx >> 2)];
  mb[p] = r ? 1 : 0;
}

__global__ void dilate2d_kernel(const unsigned char* mi, unsigned char* mo) {
  const int p = blockIdx.x * blockDim.x + threadIdx.x;
  if (p >= 1024) return;
  const int xx = p & 31, yy = p >> 5;
  unsigned char r = 0;
  for (int ky = -1; ky <= 1; ++ky)
    for (int kx = -1; kx <= 1; ++kx) {
      const int y2 = yy + ky, x2 = xx + kx;
      if ((unsigned)y2 < 32u && (unsigned)x2 < 32u) r |= mi[y2 * 32 + x2];
    }
  mo[p] = r ? 1 : 0;
}

__global__ void gn_gelu_kernel(f16* S, const float* gg, const float* gb) {
  __shared__ float red[256];
  const int g = blockIdx.x, tid = threadIdx.x;
  float sum = 0.f, sq = 0.f;
  for (int e = tid; e < 8192; e += 256) {
    const int c = g * 8 + (e >> 10), p = e & 1023;
    const float v = (float)S[(size_t)p * 256 + c];
    sum += v; sq += v * v;
  }
  red[tid] = sum; __syncthreads();
  for (int st = 128; st > 0; st >>= 1) { if (tid < st) red[tid] += red[tid + st]; __syncthreads(); }
  const float mu = red[0] / 8192.f; __syncthreads();
  red[tid] = sq; __syncthreads();
  for (int st = 128; st > 0; st >>= 1) { if (tid < st) red[tid] += red[tid + st]; __syncthreads(); }
  const float var = red[0] / 8192.f - mu * mu;
  const float inv = rsqrtf(var + 1e-5f);
  __syncthreads();
  for (int e = tid; e < 8192; e += 256) {
    const int c = g * 8 + (e >> 10), p = e & 1023;
    float v = ((float)S[(size_t)p * 256 + c] - mu) * inv * gg[c] + gb[c];
    v = 0.5f * v * (1.f + erff(v * 0.70710678f));  // exact GELU
    S[(size_t)p * 256 + c] = (f16)v;
  }
}

__global__ void head_score_kernel(const f16* __restrict__ S,
                                  const float* __restrict__ wh,
                                  const float* __restrict__ bh,
                                  float* score_ws, float* score_out) {
  const int p = blockIdx.x * blockDim.x + threadIdx.x;
  if (p >= 1024) return;
  float best = -3.4e38f;
  for (int cls = 0; cls < 10; ++cls) {
    float acc = bh[cls];
    for (int c = 0; c < 256; ++c)
      acc += (float)S[(size_t)p * 256 + c] * wh[cls * 256 + c];
    best = fmaxf(best, acc);
  }
  score_ws[p] = best;
  score_out[p] = best;
}

// full descending argsort of 1024 scores (stable: ties -> smaller index first)
__global__ void sort_kernel(const float* score, int* idx_ws, int* idx_out) {
  __shared__ float ks[1024];
  __shared__ int vs[1024];
  const int i = threadIdx.x;
  ks[i] = score[i]; vs[i] = i;
  for (int k2 = 2; k2 <= 1024; k2 <<= 1)
    for (int j = k2 >> 1; j > 0; j >>= 1) {
      __syncthreads();
      const int ixj = i ^ j;
      if (ixj > i) {
        const float a = ks[i], b = ks[ixj];
        const int va = vs[i], vb = vs[ixj];
        const bool aFirst  = (a > b) || (a == b && va < vb);
        const bool dirDesc = ((i & k2) == 0);
        if (dirDesc ? !aFirst : aFirst) {
          ks[i] = b; vs[i] = vb; ks[ixj] = a; vs[ixj] = va;
        }
      }
    }
  __syncthreads();
  idx_ws[i] = vs[i];
  idx_out[i] = vs[i];
}

__global__ void gather_kernel(const f16* __restrict__ H2, const int* __restrict__ idx,
                              float* __restrict__ pf) {
  const int e = blockIdx.x * blockDim.x + threadIdx.x;
  if (e >= 256 * 256) return;
  const int j = e >> 8, c = e & 255;
  pf[e] = (float)H2[(size_t)idx[j] * 256 + c];
}

// ---------------------------------------------------------------------------
// Host orchestration
// ---------------------------------------------------------------------------
struct Dims { int D, H, W; };

extern "C" void kernel_launch(void* const* d_in, const int* in_sizes, int n_in,
                              void* d_out, int out_size, void* d_ws, size_t ws_size,
                              hipStream_t stream) {
  (void)in_sizes; (void)n_in; (void)out_size; (void)ws_size;
  const float* vf = (const float*)d_in[0];
  const int*   vc = (const int*)d_in[1];
  int cur = 2;
  auto F = [&]() { return (const float*)d_in[cur++]; };

  size_t off = 0;  // deterministic bump allocator over d_ws
  auto alloc = [&](size_t bytes) {
    void* p = (char*)d_ws + off;
    off = (off + bytes + 255) & ~(size_t)255;
    return p;
  };
  auto lg2 = [](int v) { int s = 0; while ((1 << s) < v) ++s; return s; };

  const Dims dF{17, 256, 256}, d2{9, 128, 128}, d3{5, 64, 64},
             d4{3, 32, 32}, d5{2, 16, 16}, d6{1, 8, 8};
  auto nvox = [](Dims d) { return (size_t)d.D * d.H * d.W; };

  auto run_conv = [&](const float* w, const float* bias,
                      const float* g, const float* b, const float* m, const float* v,
                      const f16* xin, const f16* resid, const unsigned char* msk,
                      f16* yout, int CinReal, int Cinp, int Cout,
                      Dims din, Dims dout, int kd, int stride, int relu,
                      int ident_bias) {
    const int taps = kd * 9;
    const int K    = Cinp * taps;
    const int Kpad = (K + 31) & ~31;
    f16*   wk = (f16*)alloc((size_t)Kpad * Cout * sizeof(f16));
    float* s  = (float*)alloc(Cout * sizeof(float));
    float* t  = (float*)alloc(Cout * sizeof(float));
    const int tot = Kpad * Cout;
    pack_w_kernel<<<(tot + 255) / 256, 256, 0, stream>>>(w, wk, Cout, CinReal,
                                                         Cinp, taps, Kpad);
    if (ident_bias)
      ident_bias_kernel<<<(Cout + 63) / 64, 64, 0, stream>>>(bias, s, t, Cout);
    else
      fold_bn_kernel<<<(Cout + 63) / 64, 64, 0, stream>>>(g, b, m, v, bias, s, t, Cout);
    int waves = Cout / 16; if (waves > 8) waves = 8; if (waves < 1) waves = 1;
    const int M = dout.D * dout.H * dout.W;
    dim3 grid((M + 31) / 32, (Cout + 16 * waves - 1) / (16 * waves));
    conv_wmma_kernel<<<grid, 32 * waves, 0, stream>>>(
        xin, wk, s, t, resid, msk, yout,
        Cinp, lg2(Cinp), Cout, Kpad, K,
        din.D, din.H, din.W, M, lg2(dout.W), lg2(dout.H),
        kd, stride, kd == 3 ? 1 : 0, relu);
  };

  auto run_block3 = [&](f16* xin, unsigned char* msk, f16* tmp, f16* yout,
                        int C, Dims d) {
    const float* w1 = F(); const float* b1 = F();
    const float* g1 = F(); const float* bb1 = F(); const float* m1 = F(); const float* v1 = F();
    const float* w2 = F(); const float* b2 = F();
    const float* g2 = F(); const float* bb2 = F(); const float* m2 = F(); const float* v2 = F();
    run_conv(w1, b1, g1, bb1, m1, v1, xin, nullptr, msk, tmp, C, C, C, d, d, 3, 1, 1, 0);
    run_conv(w2, b2, g2, bb2, m2, v2, tmp, xin, msk, yout, C, C, C, d, d, 3, 1, 1, 0);
  };

  // ---- scatter voxels into dense f16 grid (Cin 5 padded to 8) ------------
  f16*           X0 = (f16*)alloc(nvox(dF) * 8 * sizeof(f16));
  unsigned char* M0 = (unsigned char*)alloc(nvox(dF));
  hipMemsetAsync(X0, 0, nvox(dF) * 8 * sizeof(f16), stream);
  hipMemsetAsync(M0, 0, nvox(dF), stream);
  scatter_kernel<<<(150000 + 255) / 256, 256, 0, stream>>>(vf, vc, X0, M0, 150000);

  // ---- conv_input + conv1 ------------------------------------------------
  f16* T1a = (f16*)alloc(nvox(dF) * 16 * sizeof(f16));
  f16* T1b = (f16*)alloc(nvox(dF) * 16 * sizeof(f16));
  f16* T1c = (f16*)alloc(nvox(dF) * 16 * sizeof(f16));
  {
    const float* w = F();
    const float* g = F(); const float* b = F(); const float* m = F(); const float* v = F();
    run_conv(w, nullptr, g, b, m, v, X0, nullptr, M0, T1a, 5, 8, 16, dF, dF, 3, 1, 1, 0);
  }
  run_block3(T1a, M0, T1b, T1c, 16, dF);
  run_block3(T1c, M0, T1b, T1a, 16, dF);

  // ---- stride-2 stages ----------------------------------------------------
  auto run_stage = [&](f16* xin, unsigned char* msk_in, Dims din, Dims dout,
                       int Cin, int Cout, f16** xout, unsigned char** mout) {
    unsigned char* nm = (unsigned char*)alloc(nvox(dout));
    f16* A  = (f16*)alloc(nvox(dout) * Cout * sizeof(f16));
    f16* Bb = (f16*)alloc(nvox(dout) * Cout * sizeof(f16));
    f16* Cc = (f16*)alloc(nvox(dout) * Cout * sizeof(f16));
    const int Mo = dout.D * dout.H * dout.W;
    dilate3d_kernel<<<(Mo + 255) / 256, 256, 0, stream>>>(
        msk_in, nm, din.D, din.H, din.W, dout.D, dout.H, dout.W);
    const float* w = F();
    const float* g = F(); const float* b = F(); const float* m = F(); const float* v = F();
    run_conv(w, nullptr, g, b, m, v, xin, nullptr, nm, A, Cin, Cin, Cout, din, dout, 3, 2, 1, 0);
    run_block3(A, nm, Bb, Cc, Cout, dout);
    run_block3(Cc, nm, Bb, A, Cout, dout);
    *xout = A; *mout = nm;
  };

  f16 *x2, *x3, *x4, *x5, *x6;
  unsigned char *m2, *m3, *m4, *m5, *m6;
  run_stage(T1a, M0, dF, d2, 16, 32, &x2, &m2);
  run_stage(x2, m2, d2, d3, 32, 64, &x3, &m3);
  run_stage(x3, m3, d3, d4, 64, 128, &x4, &m4);
  run_stage(x4, m4, d4, d5, 128, 128, &x5, &m5);
  run_stage(x5, m5, d5, d6, 128, 128, &x6, &m6);

  // ---- BEV collapse + 2D head --------------------------------------------
  f16*           bev = (f16*)alloc(1024 * 128 * sizeof(f16));
  unsigned char* mb  = (unsigned char*)alloc(1024);
  unsigned char* mb2 = (unsigned char*)alloc(1024);
  bev_kernel<<<(1024 * 128 + 255) / 256, 256, 0, stream>>>(x4, x5, x6, bev);
  bev_mask_kernel<<<4, 256, 0, stream>>>(m4, m5, m6, mb);
  dilate2d_kernel<<<4, 256, 0, stream>>>(mb, mb2);

  const Dims dB{1, 32, 32};
  f16* H1 = (f16*)alloc(1024 * 256 * sizeof(f16));
  f16* H2 = (f16*)alloc(1024 * 256 * sizeof(f16));
  f16* Sb = (f16*)alloc(1024 * 256 * sizeof(f16));
  {  // conv_out: SparseConv2d 3x3 128->256, BN, ReLU, dilated mask
    const float* w = F();
    const float* g = F(); const float* b = F(); const float* m = F(); const float* v = F();
    run_conv(w, nullptr, g, b, m, v, bev, nullptr, mb2, H1, 128, 128, 256, dB, dB, 1, 1, 1, 0);
  }
  {  // shared_conv: SubM 3x3 256->256 +bias, BN, ReLU, mask
    const float* w = F(); const float* b = F();
    const float* g = F(); const float* bb = F(); const float* m = F(); const float* v = F();
    run_conv(w, b, g, bb, m, v, H1, nullptr, mb2, H2, 256, 256, 256, dB, dB, 1, 1, 1, 0);
  }
  {  // sem conv 3x3 256->256 +bias (no BN/ReLU/mask), then GN(32)+GELU
    const float* w = F(); const float* b = F();
    const float* gng = F(); const float* gnb = F();
    run_conv(w, b, nullptr, nullptr, nullptr, nullptr, H2, nullptr, nullptr, Sb,
             256, 256, 256, dB, dB, 1, 1, 0, 1);
    gn_gelu_kernel<<<32, 256, 0, stream>>>(Sb, gng, gnb);
  }

  // ---- classification head, score, argsort, top-k gather ------------------
  float* score_ws = (float*)alloc(1024 * sizeof(float));
  int*   idx_ws   = (int*)alloc(1024 * sizeof(int));
  const float* wh = F(); const float* bh = F();
  float* out_f = (float*)d_out;
  head_score_kernel<<<4, 256, 0, stream>>>(Sb, wh, bh, score_ws, out_f + 65536);
  sort_kernel<<<1, 1024, 0, stream>>>(score_ws, idx_ws, (int*)d_out + 65536 + 1024);
  gather_kernel<<<(65536 + 255) / 256, 256, 0, stream>>>(H2, idx_ws, out_f);
}
